// ChamferDistance_loss_76338748719507
// MI455X (gfx1250) — matrix operand order repaired and checked by reading, not compiled, not run on hardware
//
#include <hip/hip_runtime.h>
#include <hip/hip_bf16.h>

typedef __attribute__((ext_vector_type(16))) _Float16 v16h;
typedef __attribute__((ext_vector_type(8)))  _Float16 v8h;
typedef __attribute__((ext_vector_type(8)))  float    v8f;

#define DCH    256                  // channels (K of the GEMM)
#define SPIX   96
#define NPIX   (SPIX*SPIX)          // 9216 spatial positions
#define DN     (DCH*NPIX)           // 2,359,296
#define EPSF   2.220446049250313e-16f
#define KSTEPS (DCH/32)             // 8 WMMA K-steps of 32
#define NTILES (NPIX/16)            // 576 row tiles
#define MTILES (NPIX/16)            // 576 col tiles (even -> clean unroll-by-2)
#define WAVES_PER_BLOCK 4

// ---------------------------------------------------------------------------
// Kernel 1: channel-wise L2 normalize. blockIdx.y selects X (->fp32 out + f16T)
// or Y (->fp32 ws + f16T). f16 transposed copies make WMMA fragment loads
// contiguous b128s.
// ---------------------------------------------------------------------------
__global__ void norm_kernel(const float* __restrict__ X, const float* __restrict__ Y,
                            float* __restrict__ xf_out, float* __restrict__ yf_ws,
                            _Float16* __restrict__ xfT, _Float16* __restrict__ yfT)
{
    const int n = blockIdx.x * blockDim.x + threadIdx.x;   // spatial position
    const float* in;  float* out32;  _Float16* outT;
    if (blockIdx.y == 0) { in = X; out32 = xf_out; outT = xfT; }
    else                 { in = Y; out32 = yf_ws;  outT = yfT; }

    float ss = 0.0f;
    for (int d = 0; d < DCH; ++d) {
        float v = in[d * NPIX + n];
        ss = fmaf(v, v, ss);
    }
    const float rn = 1.0f / (sqrtf(ss) + EPSF);

    for (int d0 = 0; d0 < DCH; d0 += 8) {
        v8h pack;
        #pragma unroll
        for (int j = 0; j < 8; ++j) {
            float v = in[(d0 + j) * NPIX + n] * rn;
            out32[(d0 + j) * NPIX + n] = v;
            pack[j] = (_Float16)v;
        }
        *(v8h*)(outT + (size_t)n * DCH + d0) = pack;       // 16B-aligned store
    }
}

// ---------------------------------------------------------------------------
// WMMA helpers for the fused GEMM+argmax kernel.
// ---------------------------------------------------------------------------
__device__ __forceinline__ void load_b_frag(v16h (&B)[KSTEPS],
                                            const _Float16* __restrict__ yfT,
                                            int m, int half)
{
    // B: 32x16 f16 fragment per K-step; lane holds col=m%16, K = half*8+{0..7,16..23}.
    const _Float16* bbase = yfT + (size_t)m * DCH + half * 8;
    #pragma unroll
    for (int kk = 0; kk < KSTEPS; ++kk) {
        v8h lo = *(const v8h*)(bbase + kk * 32);
        v8h hi = *(const v8h*)(bbase + kk * 32 + 16);
        #pragma unroll
        for (int j = 0; j < 8; ++j) { B[kk][j] = lo[j]; B[kk][8 + j] = hi[j]; }
    }
}

__device__ __forceinline__ void mma_update(const v16h (&A)[KSTEPS],
                                           const v16h (&B)[KSTEPS],
                                           float (&best)[8], int (&bidx)[8], int m)
{
    v8f acc = {};
    #pragma unroll
    for (int kk = 0; kk < KSTEPS; ++kk)
        acc = __builtin_amdgcn_wmma_f32_16x16x32_f16(
                  false, A[kk], false, B[kk], (short)0, acc, false, false);
    // acc[r] = sim[row(r), m]; strict '>' keeps first-max over increasing m.
    #pragma unroll
    for (int r = 0; r < 8; ++r)
        if (acc[r] > best[r]) { best[r] = acc[r]; bidx[r] = m; }
}

// ---------------------------------------------------------------------------
// Kernel 2: fused GEMM (sim = Xf^T * Yf) + row-wise argmax via WMMA.
// One wave owns a 16-row strip; A fragments (full K=256) stay in 64 VGPRs.
// m-loop is software-pipelined with two register B buffers so next-tile loads
// are in flight while the current tile's 8 WMMAs execute (low-occupancy
// dispatch => per-wave latency hiding is the dominant lever).
// ---------------------------------------------------------------------------
__global__ void __launch_bounds__(WAVES_PER_BLOCK * 32)
argmax_gemm_kernel(const _Float16* __restrict__ xfT, const _Float16* __restrict__ yfT,
                   int* __restrict__ nn_idx)
{
    const int lane = threadIdx.x & 31;
    const int wave = threadIdx.x >> 5;
    const int tile = blockIdx.x * WAVES_PER_BLOCK + wave;  // exact: grid covers 576
    const int half = lane >> 4;                            // 0 | 1
    const int sub  = lane & 15;
    const int row  = tile * 16 + sub;

    // A: 16x32 f16 fragment per K-step. Lane(L<16): K = {0..7,16..23};
    // Lane(L>=16): K = {8..15,24..31}  -> two contiguous b128 loads each.
    v16h A[KSTEPS];
    #pragma unroll
    for (int kk = 0; kk < KSTEPS; ++kk) {
        const _Float16* base = xfT + (size_t)row * DCH + kk * 32 + half * 8;
        v8h lo = *(const v8h*)(base);
        v8h hi = *(const v8h*)(base + 16);
        #pragma unroll
        for (int j = 0; j < 8; ++j) { A[kk][j] = lo[j]; A[kk][8 + j] = hi[j]; }
    }

    float best[8];
    int   bidx[8];
    #pragma unroll
    for (int r = 0; r < 8; ++r) { best[r] = -3.4e38f; bidx[r] = 0; }

    v16h B0[KSTEPS], B1[KSTEPS];
    load_b_frag(B0, yfT, 0 * 16 + sub, half);              // prologue: tile 0

    for (int mt = 0; mt < MTILES; mt += 2) {
        // Issue loads for tile mt+1 before consuming tile mt.
        load_b_frag(B1, yfT, (mt + 1) * 16 + sub, half);
        if (mt + 4 < MTILES)                               // uniform branch
            __builtin_prefetch(yfT + (size_t)((mt + 4) * 16 + sub) * DCH, 0, 3);
        mma_update(A, B0, best, bidx, mt * 16 + sub);

        // Issue loads for tile mt+2 before consuming tile mt+1.
        if (mt + 2 < MTILES)                               // uniform branch
            load_b_frag(B0, yfT, (mt + 2) * 16 + sub, half);
        mma_update(A, B1, best, bidx, (mt + 1) * 16 + sub);
    }

    // Reduce across the 16 lanes sharing each row (masks stay inside half-wave).
    #pragma unroll
    for (int r = 0; r < 8; ++r) {
        #pragma unroll
        for (int off = 8; off >= 1; off >>= 1) {
            float ov = __shfl_xor(best[r], off, 32);
            int   oi = __shfl_xor(bidx[r], off, 32);
            if (ov > best[r] || (ov == best[r] && oi < bidx[r])) {
                best[r] = ov; bidx[r] = oi;
            }
        }
    }
    if (sub == 0) {                                        // lane 0 -> rows 0..7, lane16 -> 8..15
        #pragma unroll
        for (int r = 0; r < 8; ++r)
            nn_idx[tile * 16 + r + 8 * half] = bidx[r];
    }
}

// ---------------------------------------------------------------------------
// Kernel 3: Y_sel gather + deterministic per-block squared-error partials.
// ---------------------------------------------------------------------------
__global__ void gather_loss_kernel(const float* __restrict__ xf, const float* __restrict__ yf,
                                   const int* __restrict__ nn_idx,
                                   float* __restrict__ ysel, float* __restrict__ partials)
{
    __shared__ float red[256];
    const int n   = blockIdx.x * 256 + threadIdx.x;
    const int idx = nn_idx[n];
    float acc = 0.0f;
    for (int d = 0; d < DCH; ++d) {
        float yv = yf[d * NPIX + idx];                     // gather, L2-resident
        ysel[d * NPIX + n] = yv;
        float diff = xf[d * NPIX + n] - yv;
        acc = fmaf(diff, diff, acc);
    }
    red[threadIdx.x] = acc;
    __syncthreads();
    for (int s = 128; s > 0; s >>= 1) {
        if (threadIdx.x < s) red[threadIdx.x] += red[threadIdx.x + s];
        __syncthreads();
    }
    if (threadIdx.x == 0) partials[blockIdx.x] = red[0];
}

__global__ void finalize_kernel(const float* __restrict__ partials, float* __restrict__ loss)
{
    if (threadIdx.x == 0) {
        float s = 0.0f;
        for (int i = 0; i < NPIX / 256; ++i) s += partials[i];   // deterministic order
        *loss = s / (float)DN;
    }
}

// ---------------------------------------------------------------------------
// Launch. d_out layout (tuple return order): [loss | Y_sel(DN) | Xf(DN)].
// Workspace: Yf fp32 (9.4MB) | XfT f16 (4.7MB) | YfT f16 (4.7MB) | nn_idx | partials.
// ---------------------------------------------------------------------------
extern "C" void kernel_launch(void* const* d_in, const int* in_sizes, int n_in,
                              void* d_out, int out_size, void* d_ws, size_t ws_size,
                              hipStream_t stream)
{
    (void)in_sizes; (void)n_in; (void)out_size; (void)ws_size;
    const float* X = (const float*)d_in[0];                // [1,256,96,96]
    const float* Y = (const float*)d_in[1];                // [1,256,96,96]
    // d_in[2], d_in[3] (images) are dead code in the reference — unused.

    float* out  = (float*)d_out;
    float* loss = out;
    float* ysel = out + 1;
    float* xf   = out + 1 + (size_t)DN;                    // fp32 Xf output, reused as input

    char* ws = (char*)d_ws;
    float*    yf       = (float*)ws;      ws += (size_t)DN * sizeof(float);
    _Float16* xfT      = (_Float16*)ws;   ws += (size_t)DN * sizeof(_Float16);
    _Float16* yfT      = (_Float16*)ws;   ws += (size_t)DN * sizeof(_Float16);
    int*      nn       = (int*)ws;        ws += (size_t)NPIX * sizeof(int);
    float*    partials = (float*)ws;

    dim3 gNorm(NPIX / 256, 2);
    norm_kernel<<<gNorm, 256, 0, stream>>>(X, Y, xf, yf, xfT, yfT);

    argmax_gemm_kernel<<<NTILES / WAVES_PER_BLOCK, WAVES_PER_BLOCK * 32, 0, stream>>>(
        xfT, yfT, nn);

    gather_loss_kernel<<<NPIX / 256, 256, 0, stream>>>(xf, yf, nn, ysel, partials);
    finalize_kernel<<<1, 32, 0, stream>>>(partials, loss);
}
